// DecoderTransformerBlock_71296457114378
// MI455X (gfx1250) — compile-verified, matrix operand order checked by reading, not run on hardware
//
#include <hip/hip_runtime.h>
#include <hip/hip_bf16.h>

typedef __bf16 bf16_t;
typedef __attribute__((ext_vector_type(16))) __bf16 v16bf;
typedef __attribute__((ext_vector_type(8)))  __bf16 v8bf;
typedef __attribute__((ext_vector_type(8)))  float  v8f;
typedef unsigned int        u32;
typedef unsigned long long  u64;
typedef __attribute__((ext_vector_type(4))) u32 u32x4;
typedef __attribute__((ext_vector_type(8))) u32 u32x8;

#define C_DIM   256
#define N_TOK   64
#define NH      8
#define HD      32
#define MLP_H   1024
#define B_SZ    4
#define H_SZ    160
#define W_SZ    160
#define NWW     20      // windows per row (160/8)
#define NWIN    1600    // 4 * 20 * 20
#define NTOKENS 102400  // 4*160*160
#define EPSV    1e-5f

#define V8F_ZERO {0.f,0.f,0.f,0.f,0.f,0.f,0.f,0.f}

__device__ __forceinline__ v8f wmma_bf16(v16bf a, v16bf b, v8f c) {
  // D = A(16x32 bf16) * B(32x16 bf16) + C(16x16 f32)
  return __builtin_amdgcn_wmma_f32_16x16x32_bf16(false, a, false, b, (short)0, c,
                                                 false, false);
}

// Load one 16x32 bf16 WMMA operand tile from a row-major matrix (row stride ld).
// ISA 16-bit A layout: lane = m + 16*k[3]; element e -> k = (e&7) + 8*half + 16*(e>>3).
// Per lane this is two contiguous 16-byte loads. Used for the B operand too by
// handing it the transposed (N x K) matrix.
__device__ __forceinline__ v16bf load_frag(const bf16_t* base, int ld, int row0,
                                           int k0, int lane) {
  const bf16_t* p = base + (size_t)(row0 + (lane & 15)) * ld + k0 + ((lane >> 4) << 3);
  v8bf lo = *(const v8bf*)p;
  v8bf hi = *(const v8bf*)(p + 16);
  return __builtin_shufflevector(lo, hi, 0,1,2,3,4,5,6,7,8,9,10,11,12,13,14,15);
}

// ----------------------------------------------------------- TDM (D# descr.)
// Contiguous 1-row tile: n8 elements of 8 bytes. D# per CDNA5 ISA ch.8:
// group0 = {count=1, lds_addr, global_addr[31:0], global_addr[56:32]|type=2}
// group1 = {data_size=8B, tensor_dim0=n8, tensor_dim1=1, tile_dim0=n8,
//           tile_dim1=1, tensor_dim0_stride=n8}
__device__ __forceinline__ void tdm_desc(u32x4& g0, u32x8& g1, u32 lds_addr,
                                         u64 gaddr, u32 n8) {
  u32 la  = (u32)__builtin_amdgcn_readfirstlane((int)lds_addr);
  u32 glo = (u32)__builtin_amdgcn_readfirstlane((int)(u32)gaddr);
  u32 ghi = (u32)__builtin_amdgcn_readfirstlane((int)(u32)(gaddr >> 32));
  g0[0] = 1u;                                   // count=1 valid descriptor
  g0[1] = la;                                   // LDS byte address
  g0[2] = glo;                                  // global_addr[31:0]
  g0[3] = (ghi & 0x01FFFFFFu) | 0x80000000u;    // global_addr[56:32] | type=2
  g1[0] = 0x00030000u;                          // workgroup_mask=0, data_size=8B
  g1[1] = (n8 & 0xFFFFu) << 16;                 // tensor_dim0[15:0]
  g1[2] = ((n8 >> 16) & 0xFFFFu) | 0x00010000u; // tensor_dim0[31:16], tensor_dim1=1
  g1[3] = (n8 & 0xFFFFu) << 16;                 // tile_dim0
  g1[4] = 1u;                                   // tile_dim1=1, tile_dim2=0
  g1[5] = n8;                                   // tensor_dim0_stride[31:0]
  g1[6] = 0u;
  g1[7] = 0u;
}

__device__ __forceinline__ void tdm_load_to_lds(u32 lds_addr, u64 gaddr, u32 n8) {
  u32x4 g0; u32x8 g1;
  tdm_desc(g0, g1, lds_addr, gaddr, n8);
  asm volatile("tensor_load_to_lds %0, %1" :: "s"(g0), "s"(g1) : "memory");
}

__device__ __forceinline__ void tdm_store_from_lds(u32 lds_addr, u64 gaddr, u32 n8) {
  u32x4 g0; u32x8 g1;
  tdm_desc(g0, g1, lds_addr, gaddr, n8);
  asm volatile("tensor_store_from_lds %0, %1" :: "s"(g0), "s"(g1) : "memory");
}

// Low 32 bits of the flat address of an LDS object are its LDS byte offset
// (ISA 10.2 aperture mapping).
__device__ __forceinline__ u32 lds_off(const void* p) { return (u32)(size_t)p; }

// ---------------------------------------------------------------- weight prep
// src: fp32 [in][out] (torch-Linear transposed storage) -> dst: bf16 [out][in]
__global__ __launch_bounds__(256) void convT_kernel(const float* __restrict__ src,
                                                    bf16_t* __restrict__ dst,
                                                    int in_dim, int out_dim) {
  int i = blockIdx.x * 256 + threadIdx.x;
  if (i < in_dim * out_dim) {
    int r = i / out_dim, c = i - r * out_dim;
    dst[(size_t)c * in_dim + r] = (bf16_t)src[i];
  }
}

// ------------------------------------------- LayerNorm1 + window partition
__global__ __launch_bounds__(256) void ln_part_kernel(const float* __restrict__ x,
                                                      const float* __restrict__ g,
                                                      const float* __restrict__ bta,
                                                      bf16_t* __restrict__ outp,
                                                      int do_norm) {
  __shared__ float rs[256];
  __shared__ float rq[256];
  int t = blockIdx.x, c = threadIdx.x;
  int b = t / (H_SZ * W_SZ);
  int rem = t - b * (H_SZ * W_SZ);
  int hh = rem / W_SZ;
  int ww = rem - hh * W_SZ;
  float v = x[(size_t)t * C_DIM + c];
  int win = b * (NWW * NWW) + (hh >> 3) * NWW + (ww >> 3);
  int n = ((hh & 7) << 3) + (ww & 7);
  size_t oidx = ((size_t)win * N_TOK + n) * C_DIM + c;
  if (do_norm) {
    rs[c] = v; rq[c] = v * v;
    __syncthreads();
    for (int s = 128; s > 0; s >>= 1) {
      if (c < s) { rs[c] += rs[c + s]; rq[c] += rq[c + s]; }
      __syncthreads();
    }
    float mu = rs[0] * (1.f / 256.f);
    float var = rq[0] * (1.f / 256.f) - mu * mu;
    outp[oidx] = (bf16_t)((v - mu) * rsqrtf(var + EPSV) * g[c] + bta[c]);
  } else {
    outp[oidx] = (bf16_t)v;
  }
}

// ------------------------------------------------- fused window attention
// One block = one 64-token window. 8 wave32. Everything stays in LDS.
// Tile movement global<->LDS is done by the Tensor Data Mover: each wave DMAs
// a disjoint 4KB slice (EXEC all-ones at issue; per-wave TENSORcnt wait).
__global__ __launch_bounds__(256) void attn_kernel(
    const bf16_t* __restrict__ qsrc, const bf16_t* __restrict__ kvsrc,
    const bf16_t* __restrict__ qwT, const float* __restrict__ qb,
    const bf16_t* __restrict__ kvwT, const float* __restrict__ kvb,
    const float* __restrict__ bias_tab,  // [225][8]
    const bf16_t* __restrict__ pwT, const float* __restrict__ pb,
    bf16_t* __restrict__ outp) {
  __shared__ __align__(16) bf16_t sX[N_TOK * C_DIM];     // input tile, reused as O
  __shared__ __align__(16) bf16_t sQ[N_TOK * C_DIM];     // Q, reused as proj-out
  __shared__ __align__(16) bf16_t sK[N_TOK * C_DIM];
  __shared__ __align__(16) bf16_t sVt[C_DIM * N_TOK];    // V transposed [chan][tok]
  __shared__ __align__(16) float  sS[N_TOK * N_TOK];
  __shared__ __align__(16) bf16_t sP[N_TOK * N_TOK];

  const int tid = threadIdx.x;
  const int lane = tid & 31;
  const int wave = tid >> 5;
  const int ln15 = lane & 15;
  const int half = (lane >> 4) & 1;
  const size_t wbase = (size_t)blockIdx.x * (N_TOK * C_DIM);
  const float scale = 0.17677669529663687f;  // 32^-0.5
  const u32 slice = (u32)wave * 4096u;       // per-wave 4KB DMA slice

  // ---- TDM: q-source tile (64x256 bf16 = 32KB) -> sX
  tdm_load_to_lds(lds_off(sX) + slice, (u64)(qsrc + wbase) + slice, 512u);
  __builtin_amdgcn_s_wait_tensorcnt(0);
  __syncthreads();

  // ---- Q = (X @ Wq + bq) * scale : 4x16 output tiles, 8 k-steps
  for (int j = 0; j < 8; ++j) {
    int t = wave * 8 + j, mt = t >> 4, nt = t & 15;
    v8f c = V8F_ZERO;
#pragma unroll
    for (int kk = 0; kk < 8; ++kk) {
      v16bf a = load_frag(sX, C_DIM, mt * 16, kk * 32, lane);
      v16bf b = load_frag(qwT, C_DIM, nt * 16, kk * 32, lane);
      c = wmma_bf16(a, b, c);
    }
    int col = nt * 16 + ln15;
    float bias = qb[col];
#pragma unroll
    for (int r = 0; r < 8; ++r)
      sQ[(mt * 16 + r + 8 * half) * C_DIM + col] = (bf16_t)((c[r] + bias) * scale);
  }
  __syncthreads();

  // ---- TDM: kv-source tile -> sX (overwrites; all readers passed the barrier)
  tdm_load_to_lds(lds_off(sX) + slice, (u64)(kvsrc + wbase) + slice, 512u);
  __builtin_amdgcn_s_wait_tensorcnt(0);
  __syncthreads();

  // ---- KV = X @ Wkv + bkv : cols 0..255 -> K, cols 256..511 -> V (transposed)
  for (int j = 0; j < 16; ++j) {
    int t = wave * 16 + j, mt = t >> 5, nt = t & 31;
    v8f c = V8F_ZERO;
#pragma unroll
    for (int kk = 0; kk < 8; ++kk) {
      v16bf a = load_frag(sX, C_DIM, mt * 16, kk * 32, lane);
      v16bf b = load_frag(kvwT, C_DIM, nt * 16, kk * 32, lane);
      c = wmma_bf16(a, b, c);
    }
    int col = nt * 16 + ln15;
    float bias = kvb[col];
    if (nt < 16) {
#pragma unroll
      for (int r = 0; r < 8; ++r)
        sK[(mt * 16 + r + 8 * half) * C_DIM + col] = (bf16_t)(c[r] + bias);
    } else {
      int c2 = col - C_DIM;
#pragma unroll
      for (int r = 0; r < 8; ++r)
        sVt[c2 * N_TOK + (mt * 16 + r + 8 * half)] = (bf16_t)(c[r] + bias);
    }
  }
  __syncthreads();

  bf16_t* sO = sX;  // reuse input tile as attention output accumulator

  // ---- per-head attention
  for (int h = 0; h < NH; ++h) {
    // S = Q_h @ K_h^T  (4x4 tiles, K=32 -> one WMMA each)
#pragma unroll
    for (int j = 0; j < 2; ++j) {
      int t = wave * 2 + j, mt = t >> 2, nt = t & 3;
      v8f c = V8F_ZERO;
      v16bf a = load_frag(sQ, C_DIM, mt * 16, h * HD, lane);
      v16bf b = load_frag(sK, C_DIM, nt * 16, h * HD, lane);
      c = wmma_bf16(a, b, c);
      int cc = nt * 16 + ln15;
#pragma unroll
      for (int r = 0; r < 8; ++r)
        sS[(mt * 16 + r + 8 * half) * N_TOK + cc] = c[r];
    }
    __syncthreads();

    // softmax rows with relative-position bias (computed analytically)
    if (tid < N_TOK) {
      int r = tid;
      int ih = r >> 3, iw = r & 7;
      float mx = -3.4e38f;
      for (int cc = 0; cc < N_TOK; ++cc) {
        int jh = cc >> 3, jw = cc & 7;
        int idx = (ih - jh + 7) * 15 + (iw - jw + 7);
        float s = sS[r * N_TOK + cc] + bias_tab[idx * NH + h];
        sS[r * N_TOK + cc] = s;
        mx = fmaxf(mx, s);
      }
      float sum = 0.f;
      for (int cc = 0; cc < N_TOK; ++cc) {
        float e = __expf(sS[r * N_TOK + cc] - mx);
        sS[r * N_TOK + cc] = e;
        sum += e;
      }
      float rinv = 1.f / sum;
      for (int cc = 0; cc < N_TOK; ++cc)
        sP[r * N_TOK + cc] = (bf16_t)(sS[r * N_TOK + cc] * rinv);
    }
    __syncthreads();

    // O_h = P @ V_h  (4x2 tiles, K=64 -> 2 WMMA each; one tile per wave)
    {
      int mt = wave >> 1, nt = wave & 1;
      v8f c = V8F_ZERO;
#pragma unroll
      for (int kk = 0; kk < 2; ++kk) {
        v16bf a = load_frag(sP, N_TOK, mt * 16, kk * 32, lane);
        v16bf b = load_frag(sVt, N_TOK, h * HD + nt * 16, kk * 32, lane);
        c = wmma_bf16(a, b, c);
      }
      int col = h * HD + nt * 16 + ln15;
#pragma unroll
      for (int r = 0; r < 8; ++r)
        sO[(mt * 16 + r + 8 * half) * C_DIM + col] = (bf16_t)c[r];
    }
    __syncthreads();
  }

  // ---- proj: out = O @ Wp + bp -> stage in sQ (free now), DMA out via TDM
  for (int j = 0; j < 8; ++j) {
    int t = wave * 8 + j, mt = t >> 4, nt = t & 15;
    v8f c = V8F_ZERO;
#pragma unroll
    for (int kk = 0; kk < 8; ++kk) {
      v16bf a = load_frag(sO, C_DIM, mt * 16, kk * 32, lane);
      v16bf b = load_frag(pwT, C_DIM, nt * 16, kk * 32, lane);
      c = wmma_bf16(a, b, c);
    }
    int col = nt * 16 + ln15;
    float bias = pb[col];
#pragma unroll
    for (int r = 0; r < 8; ++r)
      sQ[(mt * 16 + r + 8 * half) * C_DIM + col] = (bf16_t)(c[r] + bias);
  }
  __syncthreads();
  // per-wave 4KB DMA back to global; S_ENDPGM's implicit wait-idle covers it
  tdm_store_from_lds(lds_off(sQ) + slice, (u64)(outp + wbase) + slice, 512u);
}

// --------------------------- residual add + window reverse + LayerNorm2
__global__ __launch_bounds__(256) void resid_ln2_kernel(
    const float* __restrict__ x, const bf16_t* __restrict__ aw,
    const float* __restrict__ g, const float* __restrict__ bta,
    float* __restrict__ out, bf16_t* __restrict__ hn) {
  __shared__ float rs[256];
  __shared__ float rq[256];
  int t = blockIdx.x, c = threadIdx.x;
  int b = t / (H_SZ * W_SZ);
  int rem = t - b * (H_SZ * W_SZ);
  int hh = rem / W_SZ, ww = rem - hh * W_SZ;
  int win = b * (NWW * NWW) + (hh >> 3) * NWW + (ww >> 3);
  int n = ((hh & 7) << 3) + (ww & 7);
  float v = x[(size_t)t * C_DIM + c] +
            (float)aw[((size_t)win * N_TOK + n) * C_DIM + c];
  out[(size_t)t * C_DIM + c] = v;
  rs[c] = v; rq[c] = v * v;
  __syncthreads();
  for (int s = 128; s > 0; s >>= 1) {
    if (c < s) { rs[c] += rs[c + s]; rq[c] += rq[c + s]; }
    __syncthreads();
  }
  float mu = rs[0] * (1.f / 256.f);
  float var = rq[0] * (1.f / 256.f) - mu * mu;
  hn[(size_t)t * C_DIM + c] =
      (bf16_t)((v - mu) * rsqrtf(var + EPSV) * g[c] + bta[c]);
}

// -------------------------------------------------- MLP (fc1+GELU, fc2)
__global__ __launch_bounds__(256) void mlp_kernel(
    const bf16_t* __restrict__ hn,
    const bf16_t* __restrict__ w1T, const float* __restrict__ b1,
    const bf16_t* __restrict__ w2T, const float* __restrict__ b2,
    float* __restrict__ out) {
  __shared__ __align__(16) bf16_t sA[32 * C_DIM];   // 16 KB
  __shared__ __align__(16) bf16_t sH[32 * MLP_H];   // 64 KB
  const int tid = threadIdx.x;
  const int lane = tid & 31;
  const int wave = tid >> 5;
  const int ln15 = lane & 15;
  const int half = (lane >> 4) & 1;
  const size_t t0 = (size_t)blockIdx.x * 32;

  // TDM: 16KB activation tile -> sA (per-wave 2KB slices)
  {
    const u32 slice = (u32)wave * 2048u;
    tdm_load_to_lds(lds_off(sA) + slice, (u64)(hn + t0 * C_DIM) + slice, 256u);
    __builtin_amdgcn_s_wait_tensorcnt(0);
  }
  __syncthreads();

  // fc1 + exact GELU : 2x64 tiles, 8 k-steps
  for (int j = 0; j < 16; ++j) {
    int t = wave * 16 + j, mt = t >> 6, nt = t & 63;
    v8f c = V8F_ZERO;
#pragma unroll
    for (int kk = 0; kk < 8; ++kk) {
      v16bf a = load_frag(sA, C_DIM, mt * 16, kk * 32, lane);
      v16bf b = load_frag(w1T, C_DIM, nt * 16, kk * 32, lane);
      c = wmma_bf16(a, b, c);
    }
    int col = nt * 16 + ln15;
    float bias = b1[col];
#pragma unroll
    for (int r = 0; r < 8; ++r) {
      float v = c[r] + bias;
      v = 0.5f * v * (1.f + erff(v * 0.7071067811865475f));
      sH[(mt * 16 + r + 8 * half) * MLP_H + col] = (bf16_t)v;
    }
  }
  __syncthreads();

  // fc2 + residual accumulate into d_out : 2x16 tiles, 32 k-steps
  for (int j = 0; j < 4; ++j) {
    int t = wave * 4 + j, mt = t >> 4, nt = t & 15;
    v8f c = V8F_ZERO;
    for (int kk = 0; kk < 32; ++kk) {
      v16bf a = load_frag(sH, MLP_H, mt * 16, kk * 32, lane);
      v16bf b = load_frag(w2T, MLP_H, nt * 16, kk * 32, lane);
      c = wmma_bf16(a, b, c);
    }
    int col = nt * 16 + ln15;
    float bias = b2[col];
#pragma unroll
    for (int r = 0; r < 8; ++r) {
      size_t oi = (t0 + mt * 16 + r + 8 * half) * C_DIM + col;
      out[oi] += c[r] + bias;
    }
  }
}

// ------------------------------------------------------------------ launch
extern "C" void kernel_launch(void* const* d_in, const int* in_sizes, int n_in,
                              void* d_out, int out_size, void* d_ws, size_t ws_size,
                              hipStream_t stream) {
  (void)in_sizes; (void)n_in; (void)out_size; (void)ws_size;
  const float* x    = (const float*)d_in[0];
  const float* enc  = (const float*)d_in[1];
  // d_in[2] mask_matrix: unused by the reference forward
  const float* n1g  = (const float*)d_in[3];
  const float* n1b  = (const float*)d_in[4];
  const float* q1w  = (const float*)d_in[5];
  const float* q1b  = (const float*)d_in[6];
  const float* kv1w = (const float*)d_in[7];
  const float* kv1b = (const float*)d_in[8];
  const float* bia1 = (const float*)d_in[9];
  const float* p1w  = (const float*)d_in[10];
  const float* p1b  = (const float*)d_in[11];
  const float* q2w  = (const float*)d_in[12];
  const float* q2b  = (const float*)d_in[13];
  const float* kv2w = (const float*)d_in[14];
  const float* kv2b = (const float*)d_in[15];
  const float* bia2 = (const float*)d_in[16];
  const float* p2w  = (const float*)d_in[17];
  const float* p2b  = (const float*)d_in[18];
  const float* n2g  = (const float*)d_in[19];
  const float* n2b  = (const float*)d_in[20];
  const float* fc1w = (const float*)d_in[21];
  const float* fc1b = (const float*)d_in[22];
  const float* fc2w = (const float*)d_in[23];
  const float* fc2b = (const float*)d_in[24];
  float* out = (float*)d_out;

  char* ws = (char*)d_ws;
  // bf16 transposed weights (2 MB total)
  bf16_t* wq1t  = (bf16_t*)(ws + 0);
  bf16_t* wkv1t = (bf16_t*)(ws + 131072);
  bf16_t* wp1t  = (bf16_t*)(ws + 393216);
  bf16_t* wq2t  = (bf16_t*)(ws + 524288);
  bf16_t* wkv2t = (bf16_t*)(ws + 655360);
  bf16_t* wp2t  = (bf16_t*)(ws + 917504);
  bf16_t* wfc1t = (bf16_t*)(ws + 1048576);
  bf16_t* wfc2t = (bf16_t*)(ws + 1572864);
  // big bf16 activation buffers, 52,428,800 B each
  const size_t BIG = (size_t)NWIN * N_TOK * C_DIM * sizeof(bf16_t);
  bf16_t* bufA = (bf16_t*)(ws + 2097152);             // xw / attn2 out
  bf16_t* bufB = (bf16_t*)(ws + 2097152 + BIG);       // ew / hn
  bf16_t* bufC = (bf16_t*)(ws + 2097152 + 2 * BIG);   // attn1 out

  // 1) weight prep
  convT_kernel<<<256, 256, 0, stream>>>(q1w,  wq1t,  256, 256);
  convT_kernel<<<512, 256, 0, stream>>>(kv1w, wkv1t, 256, 512);
  convT_kernel<<<256, 256, 0, stream>>>(p1w,  wp1t,  256, 256);
  convT_kernel<<<256, 256, 0, stream>>>(q2w,  wq2t,  256, 256);
  convT_kernel<<<512, 256, 0, stream>>>(kv2w, wkv2t, 256, 512);
  convT_kernel<<<256, 256, 0, stream>>>(p2w,  wp2t,  256, 256);
  convT_kernel<<<1024, 256, 0, stream>>>(fc1w, wfc1t, 256, 1024);
  convT_kernel<<<1024, 256, 0, stream>>>(fc2w, wfc2t, 1024, 256);

  // 2) LN1 + window partition (x), partition only (enc_feat)
  ln_part_kernel<<<NTOKENS, 256, 0, stream>>>(x,   n1g, n1b, bufA, 1);
  ln_part_kernel<<<NTOKENS, 256, 0, stream>>>(enc, n1g, n1b, bufB, 0);

  // 3) windowed self-attention
  attn_kernel<<<NWIN, 256, 0, stream>>>(bufA, bufA, wq1t, q1b, wkv1t, kv1b,
                                        bia1, wp1t, p1b, bufC);
  // 4) windowed cross-attention to encoder features
  attn_kernel<<<NWIN, 256, 0, stream>>>(bufC, bufB, wq2t, q2b, wkv2t, kv2b,
                                        bia2, wp2t, p2b, bufA);

  // 5) residual + window reverse + LN2 -> d_out (x2) and hn (bf16)
  resid_ln2_kernel<<<NTOKENS, 256, 0, stream>>>(x, bufA, n2g, n2b, out, bufB);

  // 6) MLP + residual accumulate into d_out
  mlp_kernel<<<NTOKENS / 32, 256, 0, stream>>>(bufB, wfc1t, fc1b, wfc2t, fc2b, out);
}
